// DCN_5162550690698
// MI455X (gfx1250) — compile-verified, matrix operand order checked by reading, not run on hardware
//
#include <hip/hip_runtime.h>
#include <hip/hip_bf16.h>
#include <math.h>

#define B_ 4
#define C_ 256
#define H_ 64
#define W_ 64
#define O_ 256
#define K_ 9

typedef __attribute__((ext_vector_type(16))) __bf16        v16bf;
typedef __attribute__((ext_vector_type(8)))  float         v8f;
typedef __attribute__((ext_vector_type(4)))  unsigned int  uint4v;
typedef __attribute__((ext_vector_type(4)))  unsigned int  u32x4;
typedef __attribute__((ext_vector_type(8)))  int           i32x8;
typedef __attribute__((ext_vector_type(4)))  int           i32x4;

union ABFrag { uint4v u[2]; v16bf v; };
union Pack8  { __bf16 h[8]; uint4v u; };

#if __has_builtin(__builtin_amdgcn_tensor_load_to_lds) && __has_builtin(__builtin_amdgcn_s_wait_tensorcnt)
#define USE_TDM 1
#else
#define USE_TDM 0
#endif

#if USE_TDM
// Issue one TDM 2D tile load: rows x row_elems bf16 elements, row stride in elements.
// D# layout per CDNA5 ISA ch.8 (group0: count/lds/global/type, group1: dims/strides).
__device__ __forceinline__ void tdm_load_2d_bf16(
    unsigned lds_byte_addr, const unsigned short* gptr,
    int rows, int row_elems, int row_stride_elems,
    int tensor_rows, int tensor_cols)
{
  unsigned long long ga = (unsigned long long)(size_t)gptr;
  u32x4 g0 = (u32x4)0u;
  g0[0] = 1u;                                        // count=1 (valid), user mode
  g0[1] = lds_byte_addr;                             // lds_addr
  g0[2] = (unsigned)(ga & 0xffffffffull);            // global_addr[31:0]
  g0[3] = (unsigned)((ga >> 32) & 0x1ffffffull)      // global_addr[56:32]
        | (2u << 30);                                // type=2 ("image")
  i32x8 g1 = (i32x8)0;
  g1[0] = (1 << 16);                                 // wg_mask=0, data_size=1 (2 bytes)
  g1[1] = (tensor_cols & 0xffff) << 16;              // tensor_dim0[15:0]
  g1[2] = ((tensor_cols >> 16) & 0xffff)             // tensor_dim0[31:16]
        | ((tensor_rows & 0xffff) << 16);            // tensor_dim1[15:0]
  g1[3] = ((tensor_rows >> 16) & 0xffff)             // tensor_dim1[31:16]
        | ((row_elems & 0xffff) << 16);              // tile_dim0
  g1[4] = (rows & 0xffff);                           // tile_dim1 (tile_dim2=0)
  g1[5] = row_stride_elems;                          // tensor_dim0_stride[31:0]
  g1[6] = 0;                                         // stride[47:32], dim1_stride lo
  g1[7] = 0;
  i32x4 gz = (i32x4)0;                               // groups 2/3: 2D tensor -> zero
#if defined(__clang_major__) && __clang_major__ >= 23
  i32x8 gz8 = (i32x8)0;
  __builtin_amdgcn_tensor_load_to_lds(g0, g1, gz, gz, gz8, 0);
#else
  __builtin_amdgcn_tensor_load_to_lds(g0, g1, gz, gz, 0);
#endif
}
#endif

// ------------------------------------------------------------------
// Kernel 1: 3x3 conv (stride1, pad1) producing 27 channels -> offY/offX/mask
// ------------------------------------------------------------------
__global__ __launch_bounds__(256) void dcn_offset_conv(
    const float* __restrict__ x, const float* __restrict__ w_om,
    const float* __restrict__ b_om,
    float* __restrict__ offY, float* __restrict__ offX, float* __restrict__ msk)
{
  __shared__ __align__(16) float xt[16 * 18 * 18];   // 20.3 KB
  __shared__ __align__(16) float wt[27 * 16 * 9];    // 15.2 KB

  const int tid = threadIdx.x;
  const int tiles_w = W_ / 16;
  const int tiles_per_img = (H_ / 16) * tiles_w;
  const int b  = blockIdx.x / tiles_per_img;
  const int t  = blockIdx.x % tiles_per_img;
  const int h0 = (t / tiles_w) * 16;
  const int w0 = (t % tiles_w) * 16;
  const int ty = tid >> 4, tx = tid & 15;

  float acc[27];
#pragma unroll
  for (int i = 0; i < 27; ++i) acc[i] = 0.f;

  for (int c0 = 0; c0 < C_; c0 += 16) {
    __syncthreads();
    for (int idx = tid; idx < 16 * 18 * 18; idx += 256) {
      int cc = idx / 324, r = idx % 324;
      int yy = r / 18, xx = r % 18;
      int gy = h0 + yy - 1, gx = w0 + xx - 1;
      float v = 0.f;
      if (gy >= 0 && gy < H_ && gx >= 0 && gx < W_)
        v = x[(((b * C_) + (c0 + cc)) * H_ + gy) * W_ + gx];
      xt[idx] = v;
    }
    for (int idx = tid; idx < 27 * 16 * 9; idx += 256) {
      int ch = idx / 144, r = idx % 144;
      int cc = r / 9, tt = r % 9;
      wt[idx] = w_om[((ch * C_) + (c0 + cc)) * 9 + tt];
    }
    __syncthreads();

    for (int cc = 0; cc < 16; ++cc) {
#pragma unroll
      for (int tt = 0; tt < 9; ++tt) {
        float xv = xt[cc * 324 + (ty + tt / 3) * 18 + (tx + tt % 3)];
#pragma unroll
        for (int ch = 0; ch < 27; ++ch)
          acc[ch] += xv * wt[(ch * 16 + cc) * 9 + tt];
      }
    }
  }

  const int hh = h0 + ty, ww = w0 + tx;
#pragma unroll
  for (int ch = 0; ch < 27; ++ch) {
    float v = acc[ch] + b_om[ch];
    if (ch < 18) {
      int k = ch >> 1;
      float* dst = (ch & 1) ? offX : offY;
      dst[(((b * K_) + k) * H_ + hh) * W_ + ww] = v;
    } else {
      int k = ch - 18;
      msk[(((b * K_) + k) * H_ + hh) * W_ + ww] = 1.f / (1.f + expf(-v));
    }
  }
}

// ------------------------------------------------------------------
// Kernel 2: repack weight (O,C,3,3) f32 -> bf16 [k][o][c]
// ------------------------------------------------------------------
__global__ __launch_bounds__(256) void dcn_pack_w(
    const float* __restrict__ w, unsigned short* __restrict__ wB, int n)
{
  int i = blockIdx.x * 256 + threadIdx.x;
  if (i >= n) return;
  int c = i % C_;
  int o = (i / C_) % O_;
  int k = i / (C_ * O_);
  union { __bf16 b; unsigned short s; } u;
  u.b = (__bf16)w[(o * C_ + c) * K_ + k];
  wB[i] = u.s;
}

// ------------------------------------------------------------------
// Kernel 3: fused bilinear-gather + bf16 WMMA GEMM.
// Workgroup = one (b,h) row: M=64 positions x N=128 outputs, 8 waves.
// B-tile staged via Tensor Data Mover (overlaps with A-tile VALU gather).
// ------------------------------------------------------------------
__global__ __launch_bounds__(256) void dcn_main(
    const float* __restrict__ x, const unsigned short* __restrict__ wB,
    const float* __restrict__ offY, const float* __restrict__ offX,
    const float* __restrict__ msk, const float* __restrict__ bias,
    float* __restrict__ out)
{
  __shared__ __align__(16) unsigned short sA[64 * 32];   // 4 KB  (M x Kc) bf16
  __shared__ __align__(16) unsigned short sB[128 * 32];  // 8 KB  (N x Kc) bf16

  const int tid  = threadIdx.x;
  const int bh   = blockIdx.x >> 1;
  const int o0   = (blockIdx.x & 1) * 128;
  const int b    = bh / H_;
  const int h    = bh % H_;

  const int wave = tid >> 5;
  const int lane = tid & 31;
  const int hl   = lane >> 4;
  const int l16  = lane & 15;
  const int wm   = (wave >> 2) * 32;
  const int wn   = (wave & 3) * 32;

  const int pos = tid & 63;
  const int cg  = tid >> 6;

#if USE_TDM
  const unsigned sB_base = (unsigned)(size_t)(void*)sB;  // flat addr low 32 = LDS offset
#endif

  v8f acc[2][2];
#pragma unroll
  for (int i = 0; i < 2; ++i)
#pragma unroll
    for (int j = 0; j < 2; ++j)
#pragma unroll
      for (int e = 0; e < 8; ++e) acc[i][j][e] = 0.f;

  for (int k = 0; k < K_; ++k) {
    const int oidx = (((b * K_) + k) * H_ + h) * W_ + pos;
    const float oy = offY[oidx], ox = offX[oidx], mk = msk[oidx];
    const float py = (float)(h - 1 + k / 3) + oy;
    const float px = (float)(pos - 1 + k % 3) + ox;
    const float fy = floorf(py), fx = floorf(px);
    const int   y0 = (int)fy,    x0 = (int)fx;
    const float ly = py - fy,    lx = px - fx;
    const bool vy0 = (y0 >= 0) & (y0 < H_);
    const bool vy1 = (y0 + 1 >= 0) & (y0 + 1 < H_);
    const bool vx0 = (x0 >= 0) & (x0 < W_);
    const bool vx1 = (x0 + 1 >= 0) & (x0 + 1 < W_);
    const float m00 = (vy0 && vx0) ? (1.f - ly) * (1.f - lx) * mk : 0.f;
    const float m01 = (vy0 && vx1) ? (1.f - ly) * lx * mk : 0.f;
    const float m10 = (vy1 && vx0) ? ly * (1.f - lx) * mk : 0.f;
    const float m11 = (vy1 && vx1) ? ly * lx * mk : 0.f;
    const int cy0 = min(max(y0, 0), H_ - 1), cy1 = min(max(y0 + 1, 0), H_ - 1);
    const int cx0 = min(max(x0, 0), W_ - 1), cx1 = min(max(x0 + 1, 0), W_ - 1);
    const int i00 = cy0 * W_ + cx0, i01 = cy0 * W_ + cx1;
    const int i10 = cy1 * W_ + cx0, i11 = cy1 * W_ + cx1;

    for (int c0 = 0; c0 < C_; c0 += 32) {
      __syncthreads();   // all waves done reading sA/sB before restage

#if USE_TDM
      // ---- B tile via Tensor Data Mover: 128 rows x 32 bf16, row stride 256 elems.
      if (wave == 0) {
        const unsigned short* gsrc = wB + (size_t)(k * O_ + o0) * C_ + c0;
        tdm_load_2d_bf16(sB_base, gsrc, /*rows*/128, /*row_elems*/32,
                         /*row_stride*/C_, /*tensor_rows*/O_, /*tensor_cols*/C_);
      }
#else
      {
        const int n   = tid & 127;
        const int hh2 = tid >> 7;
        const uint4v* src = (const uint4v*)(wB +
            (size_t)(k * O_ + o0 + n) * C_ + c0 + hh2 * 16);
        uint4v* dst = (uint4v*)&sB[n * 32 + hh2 * 16];
        dst[0] = src[0];
        dst[1] = src[1];
        if (c0 + 32 < C_)
          __builtin_prefetch(wB + (size_t)(k * O_ + o0 + n) * C_ + c0 + 32, 0, 0);
      }
#endif

      // ---- A tile: bilinear sample 8 channels per thread (x is L2-resident);
      //      overlaps with the TDM copy above.
      {
        const float* xp = x + (size_t)((b * C_) + c0 + cg * 8) * (H_ * W_);
        Pack8 p;
#pragma unroll
        for (int cc = 0; cc < 8; ++cc) {
          const float* xc = xp + cc * (H_ * W_);
          float v = m00 * xc[i00] + m01 * xc[i01] + m10 * xc[i10] + m11 * xc[i11];
          p.h[cc] = (__bf16)v;
        }
        *(uint4v*)&sA[pos * 32 + cg * 8] = p.u;
      }

#if USE_TDM
      if (wave == 0) __builtin_amdgcn_s_wait_tensorcnt(0);  // TDM data in LDS
#endif
      __syncthreads();

      // ---- fragments (bf16 A 16x32 / B 32x16: lane half selects K-subblock)
      ABFrag a[2], bb[2];
#pragma unroll
      for (int i = 0; i < 2; ++i) {
        const int row = wm + i * 16 + l16;
        a[i].u[0] = *(const uint4v*)&sA[row * 32 + hl * 8];
        a[i].u[1] = *(const uint4v*)&sA[row * 32 + 16 + hl * 8];
      }
#pragma unroll
      for (int j = 0; j < 2; ++j) {
        const int col = wn + j * 16 + l16;
        bb[j].u[0] = *(const uint4v*)&sB[col * 32 + hl * 8];
        bb[j].u[1] = *(const uint4v*)&sB[col * 32 + 16 + hl * 8];
      }
#pragma unroll
      for (int i = 0; i < 2; ++i)
#pragma unroll
        for (int j = 0; j < 2; ++j)
          acc[i][j] = __builtin_amdgcn_wmma_f32_16x16x32_bf16(
              false, a[i].v, false, bb[j].v, (short)0, acc[i][j], false, false);
    }
  }

  // ---- epilogue: C/D layout (VGPR v <-> M=v+8*hl, N=l16) => contiguous along w
#pragma unroll
  for (int i = 0; i < 2; ++i) {
#pragma unroll
    for (int j = 0; j < 2; ++j) {
      const int o     = o0 + wn + j * 16 + l16;
      const int wbase = wm + i * 16 + hl * 8;
      const float bv  = bias[o];
      v8f r = acc[i][j];
#pragma unroll
      for (int e = 0; e < 8; ++e) r[e] += bv;
      *(v8f*)&out[(((size_t)b * O_ + o) * H_ + h) * W_ + wbase] = r;
    }
  }
}

// ------------------------------------------------------------------
extern "C" void kernel_launch(void* const* d_in, const int* in_sizes, int n_in,
                              void* d_out, int out_size, void* d_ws, size_t ws_size,
                              hipStream_t stream) {
  const float* x      = (const float*)d_in[0];
  const float* w_om   = (const float*)d_in[1];
  const float* b_om   = (const float*)d_in[2];
  const float* weight = (const float*)d_in[3];
  const float* bias   = (const float*)d_in[4];
  float* out = (float*)d_out;

  const int NOFF = B_ * K_ * H_ * W_;
  float* offY = (float*)d_ws;
  float* offX = offY + NOFF;
  float* msk  = offX + NOFF;
  unsigned short* wB = (unsigned short*)(msk + NOFF);

  dcn_offset_conv<<<B_ * (H_ / 16) * (W_ / 16), 256, 0, stream>>>(
      x, w_om, b_om, offY, offX, msk);
  const int NW = K_ * O_ * C_;
  dcn_pack_w<<<(NW + 255) / 256, 256, 0, stream>>>(weight, wB, NW);
  dcn_main<<<B_ * H_ * (O_ / 128), 256, 0, stream>>>(
      x, wB, offY, offX, msk, bias, out);
}
